// RuleLearner_24790551232963
// MI455X (gfx1250) — matrix operand order repaired and checked by reading, not compiled, not run on hardware
//
#include <hip/hip_runtime.h>
#include <hip/hip_bf16.h>

// ---------------------------------------------------------------------------
// RuleLearner forward for MI455X (gfx1250, wave32, WMMA f16 -> f32, TDM stage)
// B=16, D=64, N=H*W=256, R=32, HD=128, STEPS=5
// ---------------------------------------------------------------------------

typedef __attribute__((ext_vector_type(16))) _Float16 v16h;
typedef __attribute__((ext_vector_type(8)))  _Float16 v8h;
typedef __attribute__((ext_vector_type(8)))  float    v8f;
typedef __attribute__((ext_vector_type(4)))  unsigned int u32x4;
typedef __attribute__((ext_vector_type(8)))  int          i32x8;
typedef __attribute__((ext_vector_type(4)))  int          i32x4;

#define BB     16
#define DD     64
#define NTOK   256
#define RR     32
#define HDM    128
#define NWAVE  8
#define NSTEPS 5

#if __has_builtin(__builtin_amdgcn_tensor_load_to_lds)
#define HAVE_TDM 1
#else
#define HAVE_TDM 0
#endif

// ---- WMMA fragment helpers (CDNA5 16x16x32 f16 layouts) -------------------
// A (16x32, MxK): lane m = l%16. lanes 0-15: elems 0-7 -> K0-7, 8-15 -> K16-23.
//                 lanes 16-31: +8 on K. => two contiguous 16B loads per lane.
__device__ __forceinline__ v16h frag_a(const _Float16* p, int ld) {
  const int lane = threadIdx.x & 31;
  const _Float16* row = p + (lane & 15) * ld + ((lane >> 4) << 3);
  v8h lo = *(const v8h*)(row);
  v8h hi = *(const v8h*)(row + 16);
  return __builtin_shufflevector(lo, hi, 0,1,2,3,4,5,6,7,8,9,10,11,12,13,14,15);
}

// B (32x16, KxN) from n-major storage stor[n][k] (ld = k-stride of an n-row):
// lane n = l%16; lanes 0-15: elems 0-7 -> K0-7, 8-15 -> K16-23; lanes 16-31: +8 on K.
__device__ __forceinline__ v16h frag_b(const _Float16* p, int ld) {
  const int lane = threadIdx.x & 31;
  const _Float16* row = p + (lane & 15) * ld + ((lane >> 4) << 3);
  v8h lo = *(const v8h*)(row);
  v8h hi = *(const v8h*)(row + 16);
  return __builtin_shufflevector(lo, hi, 0,1,2,3,4,5,6,7,8,9,10,11,12,13,14,15);
}

__device__ __forceinline__ v8f wmma16(v16h a, v16h b, v8f c) {
  return __builtin_amdgcn_wmma_f32_16x16x32_f16(false, a, false, b, (short)0, c,
                                                false, false);
}

__device__ __forceinline__ void wait_tensorcnt0() {
#if __has_builtin(__builtin_amdgcn_s_wait_tensorcnt)
  __builtin_amdgcn_s_wait_tensorcnt(0);
#else
  asm volatile("s_wait_tensorcnt 0x0" ::: "memory");
#endif
}

// ---- setup / epilogue kernels ---------------------------------------------
__global__ void init_x_kernel(const float* __restrict__ symbols,
                              float* __restrict__ x_acc) {
  int i = blockIdx.x * 256 + threadIdx.x;          // over B*N*D
  if (i >= BB * NTOK * DD) return;
  int d = i & (DD - 1);
  int n = (i >> 6) & (NTOK - 1);
  int b = i >> 14;
  x_acc[i] = symbols[(b * DD + d) * NTOK + n];     // (B,D,N) -> (B,N,D)
}

__global__ void conv_w_kernel(const float* __restrict__ Wq, const float* __restrict__ Wk,
                              const float* __restrict__ W1, const float* __restrict__ W2,
                              _Float16* __restrict__ WqT, _Float16* __restrict__ WkT,
                              _Float16* __restrict__ W1T, _Float16* __restrict__ W2T) {
  int i = blockIdx.x * 256 + threadIdx.x;          // over R*HD*D, i=[r][h][d]
  if (i >= RR * HDM * DD) return;
  int d = i & (DD - 1);
  int h = (i >> 6) & (HDM - 1);
  int r = i >> 13;
  WqT[i] = (_Float16)Wq[(r * DD + d) * HDM + h];   // [r][h][d] <- (R,D,HD)
  WkT[i] = (_Float16)Wk[(r * DD + d) * HDM + h];
  W1T[i] = (_Float16)W1[(r * DD + d) * HDM + h];
  W2T[(r * DD + d) * HDM + h] = (_Float16)W2[(r * HDM + h) * DD + d]; // [r][d][h]
}

__global__ void x_to_f16_kernel(const float* __restrict__ x_acc,
                                _Float16* __restrict__ xf16) {
  int i = blockIdx.x * 256 + threadIdx.x;
  if (i < BB * NTOK * DD) xf16[i] = (_Float16)x_acc[i];
}

__global__ void write_out_kernel(const float* __restrict__ x_acc,
                                 float* __restrict__ out) {
  int i = blockIdx.x * 256 + threadIdx.x;          // over (B,D,N) output
  if (i >= BB * DD * NTOK) return;
  int n = i & (NTOK - 1);
  int d = (i >> 8) & (DD - 1);
  int b = i >> 14;
  out[i] = x_acc[(b * NTOK + n) * DD + d];
}

// ---- gate MLP + softmax over rules ----------------------------------------
__global__ __launch_bounds__(128) void gate_kernel(
    const float* __restrict__ x_acc, const float* __restrict__ Wg1,
    const float* __restrict__ bg1, const float* __restrict__ Wg2,
    const float* __restrict__ bg2, float* __restrict__ g) {
  __shared__ float pooled[DD];
  __shared__ float hbuf[HDM];
  __shared__ float lg[RR];
  const int b = blockIdx.x;
  const int t = threadIdx.x;
  if (t < DD) {
    float acc = 0.f;
    for (int n = 0; n < NTOK; ++n) acc += x_acc[(b * NTOK + n) * DD + t];
    pooled[t] = acc * (1.0f / NTOK);
  }
  __syncthreads();
  {
    float acc = bg1[t];
    for (int d = 0; d < DD; ++d) acc += pooled[d] * Wg1[d * HDM + t];
    hbuf[t] = fmaxf(acc, 0.f);
  }
  __syncthreads();
  if (t < RR) {
    float acc = bg2[t];
    for (int i = 0; i < HDM; ++i) acc += hbuf[i] * Wg2[i * RR + t];
    lg[t] = acc;
  }
  __syncthreads();
  if (t == 0) {
    float mx = lg[0];
    for (int i = 1; i < RR; ++i) mx = fmaxf(mx, lg[i]);
    float sum = 0.f;
    for (int i = 0; i < RR; ++i) { lg[i] = __expf(lg[i] - mx); sum += lg[i]; }
    float inv = 1.f / sum;
    for (int i = 0; i < RR; ++i) g[b * RR + i] = lg[i] * inv;
  }
}

// ---- main rule kernel: one block per (b, r) -------------------------------
__global__ __launch_bounds__(256, 1) void rule_kernel(
    const _Float16* __restrict__ xf16, const _Float16* __restrict__ WqT,
    const _Float16* __restrict__ WkT, const _Float16* __restrict__ W1T,
    const _Float16* __restrict__ W2T, const float* __restrict__ bq,
    const float* __restrict__ bk, const float* __restrict__ b1,
    const float* __restrict__ b2, const float* __restrict__ g,
    float* __restrict__ x_acc) {
  extern __shared__ char smem_raw[];
  _Float16* xs  = (_Float16*)smem_raw;       // [NTOK][DD]   32 KB  (LDS offset 0)
  _Float16* xsT = xs  + NTOK * DD;           // [DD][NTOK]   32 KB
  _Float16* qs  = xsT + DD * NTOK;           // [NTOK][HDM]  64 KB
  _Float16* ks  = qs  + NTOK * HDM;          // [NTOK][HDM]  64 KB
  _Float16* pw  = ks  + NTOK * HDM;          // [NWAVE][16][NTOK] 64 KB

  const int b    = blockIdx.x / RR;
  const int r    = blockIdx.x % RR;
  const int wave = threadIdx.x >> 5;
  const int lane = threadIdx.x & 31;
  const int nlo  = lane & 15;
  const int mhi  = (lane >> 4) << 3;         // C-layout row offset (0 or 8)

#if HAVE_TDM
  // ---- stage x tile via Tensor Data Mover: one 32KB contiguous row ----
  if (wave == 0) {
    const unsigned long long ga =
        (unsigned long long)(const void*)(xf16 + (size_t)b * NTOK * DD);
    u32x4 g0;
    g0[0] = 1u;                                   // count=1, user descriptor
    g0[1] = 0u;                                   // lds_addr = 0 (xs @ LDS base)
    g0[2] = (unsigned)(ga & 0xffffffffu);         // global_addr[31:0]
    g0[3] = (unsigned)((ga >> 32) & 0x01ffffffu)  // global_addr[56:32]
            | 0x80000000u;                        // type = 2 ("image")
    i32x8 g1;
    g1[0] = 0x00010000;     // workgroup_mask=0, data_size=1 (2 bytes/elem)
    g1[1] = 0x40000000;     // tensor_dim0 = 16384 (low16 -> bits 63:48)
    g1[2] = 0x00010000;     // tensor_dim0 hi=0; tensor_dim1 = 1
    g1[3] = 0x40000000;     // tensor_dim1 hi=0; tile_dim0 = 16384
    g1[4] = 0x00000001;     // tile_dim1 = 1, tile_dim2 = 0
    g1[5] = 16384;          // tensor_dim0_stride[31:0]
    g1[6] = 0;              // stride hi, tensor_dim1_stride lo
    g1[7] = 0;
    i32x4 z4 = {0, 0, 0, 0};
#if defined(__clang_major__) && (__clang_major__ >= 23)
    i32x8 z8 = {0, 0, 0, 0, 0, 0, 0, 0};
    __builtin_amdgcn_tensor_load_to_lds(g0, g1, z4, z4, z8, 0);
#else
    __builtin_amdgcn_tensor_load_to_lds(g0, g1, z4, z4, 0);
#endif
    wait_tensorcnt0();
  }
  __syncthreads();
  for (int i = threadIdx.x; i < NTOK * DD; i += 256) {
    _Float16 h = xs[i];
    xsT[(i & (DD - 1)) * NTOK + (i >> 6)] = h;    // build transposed copy
  }
  __syncthreads();
#else
  // ---- fallback: stage state tile (row-major + transposed) into LDS ----
  for (int i = threadIdx.x; i < NTOK * DD; i += 256) {
    _Float16 h = xf16[b * NTOK * DD + i];
    xs[i] = h;
    xsT[(i & (DD - 1)) * NTOK + (i >> 6)] = h;
  }
  __syncthreads();
#endif

  // ---- Q = x*Wq + bq, K = x*Wk + bk (cooperative, 16x8 tiles each) ----
  const _Float16* wq = WqT + r * HDM * DD;
  const _Float16* wk = WkT + r * HDM * DD;
  __builtin_prefetch(wq, 0, 1);
  __builtin_prefetch(wk, 0, 1);
  for (int job = wave; job < 128; job += NWAVE) {
    const int tm = job >> 3, tn = job & 7;
    v8f cq = {}; v8f ck = {};
#pragma unroll
    for (int kk = 0; kk < DD; kk += 32) {
      v16h a = frag_a(xs + tm * 16 * DD + kk, DD);
      cq = wmma16(a, frag_b(wq + tn * 16 * DD + kk, DD), cq);
      ck = wmma16(a, frag_b(wk + tn * 16 * DD + kk, DD), ck);
    }
    const float bqv = bq[r * HDM + tn * 16 + nlo];
    const float bkv = bk[r * HDM + tn * 16 + nlo];
#pragma unroll
    for (int j = 0; j < 8; ++j) {
      qs[(tm * 16 + mhi + j) * HDM + tn * 16 + nlo] = (_Float16)(cq[j] + bqv);
      ks[(tm * 16 + mhi + j) * HDM + tn * 16 + nlo] = (_Float16)(ck[j] + bkv);
    }
  }
  __syncthreads();

  const float gval  = g[b * RR + r];
  const float scale = 0.08838834764831845f;  // 1/sqrt(128)
  _Float16* myp = pw + wave * 16 * NTOK;     // wave-private 16x256 strip

  for (int t = 0; t < 2; ++t) {              // two 16-row query tiles per wave
    const int q0 = (wave * 2 + t) * 16;

    // ---- scores S = Q K^T (16 x 256 held in registers) ----
    v8f s[16];
#pragma unroll
    for (int kt = 0; kt < 16; ++kt) {
      v8f c = {};
#pragma unroll
      for (int kk = 0; kk < HDM; kk += 32) {
        v16h a  = frag_a(qs + q0 * HDM + kk, HDM);
        v16h bf = frag_b(ks + kt * 16 * HDM + kk, HDM);
        c = wmma16(a, bf, c);
      }
      s[kt] = c;
    }

    // ---- row softmax (rows split across 16-lane halves per C layout) ----
#pragma unroll
    for (int j = 0; j < 8; ++j) {
      float mx = -1e30f;
#pragma unroll
      for (int kt = 0; kt < 16; ++kt) mx = fmaxf(mx, s[kt][j] * scale);
#pragma unroll
      for (int msk = 8; msk >= 1; msk >>= 1) mx = fmaxf(mx, __shfl_xor(mx, msk, 32));
      float sum = 0.f;
#pragma unroll
      for (int kt = 0; kt < 16; ++kt) {
        float e = __expf(s[kt][j] * scale - mx);
        s[kt][j] = e;
        sum += e;
      }
#pragma unroll
      for (int msk = 8; msk >= 1; msk >>= 1) sum += __shfl_xor(sum, msk, 32);
      const float inv = 1.0f / sum;
#pragma unroll
      for (int kt = 0; kt < 16; ++kt)
        myp[(mhi + j) * NTOK + kt * 16 + nlo] = (_Float16)(s[kt][j] * inv);
    }

    // ---- ctx = P * x  (16 x 64) ----
    v8f ctx[4] = {};
#pragma unroll
    for (int kc = 0; kc < NTOK; kc += 32) {
      v16h a = frag_a(myp + kc, NTOK);
#pragma unroll
      for (int dt = 0; dt < 4; ++dt)
        ctx[dt] = wmma16(a, frag_b(xsT + dt * 16 * NTOK + kc, NTOK), ctx[dt]);
    }

    // ---- FFN: t = relu(ctx W1 + b1) W2 + b2, gated atomic accumulate ----
    _Float16* cts = myp;                 // reuse strip: [16][DD]
    _Float16* hs  = myp + 16 * DD;       // [16][HDM]
#pragma unroll
    for (int dt = 0; dt < 4; ++dt)
#pragma unroll
      for (int j = 0; j < 8; ++j)
        cts[(mhi + j) * DD + dt * 16 + nlo] = (_Float16)ctx[dt][j];

    const _Float16* w1 = W1T + r * HDM * DD;
#pragma unroll
    for (int nt = 0; nt < 8; ++nt) {
      v8f c = {};
#pragma unroll
      for (int kk = 0; kk < DD; kk += 32) {
        v16h a = frag_a(cts + kk, DD);
        c = wmma16(a, frag_b(w1 + nt * 16 * DD + kk, DD), c);
      }
      const float bv = b1[r * HDM + nt * 16 + nlo];
#pragma unroll
      for (int j = 0; j < 8; ++j)
        hs[(mhi + j) * HDM + nt * 16 + nlo] = (_Float16)fmaxf(c[j] + bv, 0.f);
    }

    const _Float16* w2 = W2T + r * DD * HDM;
#pragma unroll
    for (int dt = 0; dt < 4; ++dt) {
      v8f c = {};
#pragma unroll
      for (int kk = 0; kk < HDM; kk += 32) {
        v16h a = frag_a(hs + kk, HDM);
        c = wmma16(a, frag_b(w2 + dt * 16 * HDM + kk, HDM), c);
      }
      const float bv = b2[r * DD + dt * 16 + nlo];
#pragma unroll
      for (int j = 0; j < 8; ++j) {
        const int row = q0 + mhi + j;
        unsafeAtomicAdd(&x_acc[(b * NTOK + row) * DD + dt * 16 + nlo],
                        gval * (c[j] + bv));
      }
    }
  }
}

// ---------------------------------------------------------------------------
extern "C" void kernel_launch(void* const* d_in, const int* in_sizes, int n_in,
                              void* d_out, int out_size, void* d_ws, size_t ws_size,
                              hipStream_t stream) {
  const float* symbols = (const float*)d_in[0];
  const float* Wq  = (const float*)d_in[1];
  const float* bq  = (const float*)d_in[2];
  const float* Wk  = (const float*)d_in[3];
  const float* bk  = (const float*)d_in[4];
  const float* W1  = (const float*)d_in[5];
  const float* b1  = (const float*)d_in[6];
  const float* W2  = (const float*)d_in[7];
  const float* b2  = (const float*)d_in[8];
  // d_in[9..12] = confidence params (unused in forward)
  const float* Wg1 = (const float*)d_in[13];
  const float* bg1 = (const float*)d_in[14];
  const float* Wg2 = (const float*)d_in[15];
  const float* bg2 = (const float*)d_in[16];

  const int NE = BB * NTOK * DD;             // 262144 state elements
  const size_t WSZ = (size_t)RR * HDM * DD;  // 262144 weight elements per tensor

  char* ws = (char*)d_ws;
  size_t off = 0;
  float*    x_acc = (float*)(ws + off);    off += (size_t)NE * sizeof(float);
  _Float16* xf16  = (_Float16*)(ws + off); off += (size_t)NE * sizeof(_Float16);
  _Float16* WqT   = (_Float16*)(ws + off); off += WSZ * sizeof(_Float16);
  _Float16* WkT   = (_Float16*)(ws + off); off += WSZ * sizeof(_Float16);
  _Float16* W1T   = (_Float16*)(ws + off); off += WSZ * sizeof(_Float16);
  _Float16* W2T   = (_Float16*)(ws + off); off += WSZ * sizeof(_Float16);
  float*    gbuf  = (float*)(ws + off);    off += (size_t)BB * RR * sizeof(float);

  const size_t lds_bytes =
      ((size_t)NTOK * DD * 2 + (size_t)NTOK * HDM * 2 + (size_t)NWAVE * 16 * NTOK) *
      sizeof(_Float16);                      // 256 KB
  hipFuncSetAttribute(reinterpret_cast<const void*>(rule_kernel),
                      hipFuncAttributeMaxDynamicSharedMemorySize,
                      (int)lds_bytes);

  init_x_kernel<<<NE / 256, 256, 0, stream>>>(symbols, x_acc);
  conv_w_kernel<<<(int)(WSZ / 256), 256, 0, stream>>>(Wq, Wk, W1, W2,
                                                      WqT, WkT, W1T, W2T);

  for (int step = 0; step < NSTEPS; ++step) {
    gate_kernel<<<BB, 128, 0, stream>>>(x_acc, Wg1, bg1, Wg2, bg2, gbuf);
    x_to_f16_kernel<<<NE / 256, 256, 0, stream>>>(x_acc, xf16);
    rule_kernel<<<BB * RR, 256, lds_bytes, stream>>>(
        xf16, WqT, WkT, W1T, W2T, bq, bk, b1, b2, gbuf, x_acc);
  }

  write_out_kernel<<<NE / 256, 256, 0, stream>>>(x_acc, (float*)d_out);
}
